// Attention_89103391523461
// MI455X (gfx1250) — matrix operand order repaired and checked by reading, not compiled, not run on hardware
//
#include <hip/hip_runtime.h>
#include <hip/hip_bf16.h>

#define Bq 2
#define Nq 2048
#define DIMq 1024
#define Hq 16
#define DHq 64
#define Mq 16
#define Jq (Nq + Mq)      // 2064
#define TOPKq 64
#define SCALEq 0.125f     // 64^-0.5
#define MASKVAL (-3.402823466e38f)

typedef __attribute__((ext_vector_type(16))) _Float16 v16h;
typedef __attribute__((ext_vector_type(8)))  _Float16 v8h;
typedef __attribute__((ext_vector_type(8)))  float    v8f;
typedef unsigned int v4u __attribute__((ext_vector_type(4)));
typedef int v4i __attribute__((ext_vector_type(4)));
typedef int v8i __attribute__((ext_vector_type(8)));

// ---------------- WMMA fragment helpers (contiguous k = 16*(lane/16) + e order, wave32) ----------------

// A-matrix 16x32 f16, row-major source.
// lanes 0-15 / 16-31 both cover rows 0-15; low lanes hold K 0-15, high lanes K 16-31.
static __device__ inline v16h load_a16(const _Float16* __restrict__ base, int ld, int k0) {
  const int lane = threadIdx.x & 31;
  const _Float16* p = base + (size_t)(lane & 15) * ld + (k0 + ((lane >> 4) << 4));
  v8h lo = *reinterpret_cast<const v8h*>(p);
  v8h hi = *reinterpret_cast<const v8h*>(p + 8);
  v16h a;
#pragma unroll
  for (int e = 0; e < 8; ++e) { a[e] = lo[e]; a[e + 8] = hi[e]; }
  return a;
}

// B-matrix 32x16 f16 from column-major (transposed) storage: lane&15 = column,
// lanes 0-15 hold K 0-15, lanes 16-31 hold K 16-31 (contiguous 32B load).
// Works for global or LDS (__shared__) source pointers.
static __device__ inline v16h load_b16(const _Float16* baseT, int ldT, int k0) {
  const int lane = threadIdx.x & 31;
  const _Float16* p = baseT + (size_t)(lane & 15) * ldT + (k0 + ((lane >> 4) << 4));
  return *reinterpret_cast<const v16h*>(p);
}

static __device__ inline v16h load_a16_guard(const _Float16* __restrict__ base, int ld, int k0, int klim) {
  const int lane = threadIdx.x & 31;
  const _Float16* p = base + (size_t)(lane & 15) * ld;
  const int kb = k0 + ((lane >> 4) << 4);
  v16h a;
#pragma unroll
  for (int e = 0; e < 8; ++e) {
    a[e]     = (kb + e      < klim) ? p[kb + e]      : (_Float16)0.f;
    a[e + 8] = (kb + 8 + e < klim) ? p[kb + 8 + e] : (_Float16)0.f;
  }
  return a;
}

static __device__ inline v16h load_b16_guard(const _Float16* __restrict__ baseT, int ldT, int k0, int klim) {
  const int lane = threadIdx.x & 31;
  const _Float16* p = baseT + (size_t)(lane & 15) * ldT;
  const int kb = k0 + ((lane >> 4) << 4);
  v16h b;
#pragma unroll
  for (int e = 0; e < 16; ++e) b[e] = (kb + e < klim) ? p[kb + e] : (_Float16)0.f;
  return b;
}

static __device__ inline v8f wmma32(v16h a, v16h b, v8f c) {
  // 8 args: (neg_a, A, neg_b, B, c_mod, C, reuse_a, reuse_b)
  return __builtin_amdgcn_wmma_f32_16x16x32_f16(false, a, false, b, (short)0, c, false, false);
}

// ---------------- TDM: DMA one [rows x 64] f16 row-major tile from global to LDS ----------
// Descriptor per CDNA5 ISA ch.8 (D# groups). 2D tile, data_size=2B, dim0=stride=64.
// All descriptor fields are wave-uniform (TDM sources the D# from SGPRs).
// This toolchain exposes the 6-arg builtin: (v4u, v8i, v4i, v4i, v8i, i32 cpol).
static __device__ inline void tdm_load_tile64(const _Float16* gsrc, _Float16* lds_dst, int rows) {
  unsigned long long ga = (unsigned long long)(uintptr_t)gsrc;
  unsigned lds_addr = (unsigned)(uintptr_t)lds_dst;  // low 32 bits of generic ptr = LDS byte addr
  v4u g0;
  g0[0] = 1u;                                         // count=1, user descriptor
  g0[1] = lds_addr;                                   // lds_addr (bytes)
  g0[2] = (unsigned)(ga & 0xffffffffu);               // global_addr[31:0]
  g0[3] = (unsigned)((ga >> 32) & 0x01ffffffu) | 0x80000000u;  // global_addr[56:32], type=2
  v8i g1;
  g1[0] = 0x00010000;                 // data_size=2B (code 1), no multicast/iterate/pad
  g1[1] = (int)(64u << 16);           // tensor_dim0[15:0]=64 at bits 63:48
  g1[2] = (int)((unsigned)rows << 16);// tensor_dim1[15:0]=rows at bits 95:80
  g1[3] = (int)(64u << 16);           // tile_dim0=64 at bits 127:112
  g1[4] = rows;                       // tile_dim1=rows at bits 143:128 (tile_dim2=0)
  g1[5] = 64;                         // tensor_dim0_stride[31:0]=64
  g1[6] = 0;                          // stride hi / dim1_stride lo
  g1[7] = 0;
  v4i gz = {0, 0, 0, 0};              // groups 2/3 unused (2D tensor)
  v8i gz8 = {0, 0, 0, 0, 0, 0, 0, 0};
  __builtin_amdgcn_tensor_load_to_lds(g0, g1, gz, gz, gz8, 0);
}

// ---------------- wave32 reductions ----------------
static __device__ inline int wred_add_i(int v) {
#pragma unroll
  for (int m = 16; m >= 1; m >>= 1) v += __shfl_xor(v, m, 32);
  return v;
}
static __device__ inline float wred_add_f(float v) {
#pragma unroll
  for (int m = 16; m >= 1; m >>= 1) v += __shfl_xor(v, m, 32);
  return v;
}
static __device__ inline unsigned wred_max_u(unsigned v) {
#pragma unroll
  for (int m = 16; m >= 1; m >>= 1) {
    unsigned o = (unsigned)__shfl_xor((int)v, m, 32);
    v = (o > v) ? o : v;
  }
  return v;
}

// order-preserving float<->uint key transform
static __device__ inline unsigned f2k(float f) {
  unsigned u = __float_as_uint(f);
  return u ^ (unsigned)(((int)u >> 31) | (int)0x80000000);
}
static __device__ inline float k2f(unsigned k) {
  unsigned u = (k & 0x80000000u) ? (k ^ 0x80000000u) : ~k;
  return __uint_as_float(u);
}

// ---------------- Kernel 0: conversions / packing ----------------

__global__ void k_cvt(const float* __restrict__ src, _Float16* __restrict__ dst, int n) {
  int i = blockIdx.x * blockDim.x + threadIdx.x;
  if (i < n) dst[i] = (_Float16)src[i];
}

// WqkvT[c][k], c in [0,3072): cols 0-1023 from Wq, 1024-3071 from Wkv
__global__ void k_pack_wqkv(const float* __restrict__ Wq, const float* __restrict__ Wkv,
                            _Float16* __restrict__ WqkvT) {
  int i = blockIdx.x * blockDim.x + threadIdx.x;
  int c = i >> 10, k = i & 1023;
  float v = (c < 1024) ? Wq[(size_t)k * 1024 + c] : Wkv[(size_t)k * 2048 + (c - 1024)];
  WqkvT[(size_t)c * 1024 + k] = (_Float16)v;
}

__global__ void k_pack_wout(const float* __restrict__ Wout, _Float16* __restrict__ WoutT) {
  int i = blockIdx.x * blockDim.x + threadIdx.x;
  int c = i >> 10, k = i & 1023;
  WoutT[(size_t)c * 1024 + k] = (_Float16)Wout[(size_t)k * 1024 + c];
}

// prepend mem kv: k_h[b][h][0..M-1][d], v_t[b][h][d][0..M-1]
__global__ void k_init_mem(const float* __restrict__ mem_k, const float* __restrict__ mem_v,
                           _Float16* __restrict__ k_h, _Float16* __restrict__ v_t) {
  int i = blockIdx.x * blockDim.x + threadIdx.x;  // < B*H*M*DH
  int d = i & 63, m = (i >> 6) & 15, h = (i >> 10) & 15, b = i >> 14;
  float kv = mem_k[((size_t)h * Mq + m) * DHq + d];
  float vv = mem_v[((size_t)h * Mq + m) * DHq + d];
  k_h[(((size_t)(b * Hq + h) * Jq) + m) * DHq + d] = (_Float16)kv;
  v_t[(((size_t)(b * Hq + h) * DHq) + d) * Jq + m] = (_Float16)vv;
}

// ---------------- Kernel 1: fused QKV projection GEMM (WMMA) ----------------
// One wave computes a 32x64 tile of [BN x 3072] = x_h @ WqkvT^T (B fragments reused by 2 row-frags),
// scatters into q/k/v layouts.
__global__ void k_gemm_qkv(const _Float16* __restrict__ x_h, const _Float16* __restrict__ WqkvT,
                           _Float16* __restrict__ q_h, _Float16* __restrict__ k_h,
                           _Float16* __restrict__ v_t) {
  const int lane = threadIdx.x & 31;
  const int row0 = blockIdx.y << 5;   // row in [0, B*N), 32 rows per wave
  const int col0 = blockIdx.x << 6;   // col in [0, 3072)
  const _Float16* A0 = x_h + (size_t)row0 * DIMq;
  const _Float16* A1 = A0 + (size_t)16 * DIMq;
  v8f acc[2][4] = {};
  for (int k0 = 0; k0 < DIMq; k0 += 32) {
    if (k0 + 32 < DIMq)
      __builtin_prefetch(A0 + (size_t)(lane & 15) * DIMq + k0 + 32, 0, 1);
    v16h a0 = load_a16(A0, DIMq, k0);
    v16h a1 = load_a16(A1, DIMq, k0);
#pragma unroll
    for (int t = 0; t < 4; ++t) {
      v16h b = load_b16(WqkvT + (size_t)(col0 + 16 * t) * DIMq, DIMq, k0);
      acc[0][t] = wmma32(a0, b, acc[0][t]);
      acc[1][t] = wmma32(a1, b, acc[1][t]);
    }
  }
  const int rl = (lane >> 4) << 3, cl = lane & 15;
#pragma unroll
  for (int rf = 0; rf < 2; ++rf) {
#pragma unroll
    for (int t = 0; t < 4; ++t) {
#pragma unroll
      for (int v = 0; v < 8; ++v) {
        int row = row0 + 16 * rf + rl + v;
        int col = col0 + 16 * t + cl;
        int b = row >> 11;       // / N
        int n = row & (Nq - 1);
        _Float16 hv = (_Float16)acc[rf][t][v];
        if (col < 1024) {
          int h = col >> 6, d = col & 63;
          q_h[(((size_t)(b * Hq + h) * Nq + n) * DHq) + d] = hv;
        } else if (col < 2048) {
          int c = col - 1024; int h = c >> 6, d = c & 63;
          k_h[(((size_t)(b * Hq + h) * Jq + (Mq + n)) * DHq) + d] = hv;
        } else {
          int c = col - 2048; int h = c >> 6, d = c & 63;
          v_t[(((size_t)(b * Hq + h) * DHq + d) * Jq) + (Mq + n)] = hv;
        }
      }
    }
  }
}

// ---------------- Kernel 2: QK^T dots + talking-heads mix + scale + causal mask ----------------
// Workgroup (256 thr = 8 waves) per (b, i-tile16, j-tile16). Each wave handles 2 heads:
// K-tiles are DMA'd to LDS by the Tensor Data Mover, B fragments then come from LDS.
__global__ void k_dots_mix(const _Float16* __restrict__ q_h, const _Float16* __restrict__ k_h,
                           const float* __restrict__ pre_proj, float* __restrict__ Dmix) {
  __shared__ __align__(32) _Float16 Ktile[Hq][16][DHq];  // 32 KB, [head][j][d]
  __shared__ float S[Hq][16][17];                        // [input head][i][j], padded
  __shared__ float Pp[Hq][Hq];                           // pre_proj[h][kk]
  const int b = blockIdx.z, i0 = blockIdx.y << 4, j0 = blockIdx.x << 4;
  const int tid = threadIdx.x, lane = tid & 31, wave = tid >> 5;
  Pp[tid >> 4][tid & 15] = pre_proj[tid];

  // issue TDM loads for this wave's two K-tiles (descriptors are wave-uniform)
#pragma unroll
  for (int u = 0; u < 2; ++u) {
    const int h = wave * 2 + u;
    tdm_load_tile64(k_h + ((size_t)(b * Hq + h) * Jq + j0) * DHq, &Ktile[h][0][0], 16);
  }
  __builtin_amdgcn_s_wait_tensorcnt(0);

#pragma unroll
  for (int u = 0; u < 2; ++u) {
    const int h = wave * 2 + u;
    const _Float16* Aq = q_h + ((size_t)(b * Hq + h) * Nq + i0) * DHq;
    const _Float16* Bk = &Ktile[h][0][0];   // LDS, col-major over j, ld=DH
    v8f acc = {};
    acc = wmma32(load_a16(Aq, DHq, 0),  load_b16(Bk, DHq, 0),  acc);
    acc = wmma32(load_a16(Aq, DHq, 32), load_b16(Bk, DHq, 32), acc);
    const int rl = (lane >> 4) << 3, cl = lane & 15;
#pragma unroll
    for (int v = 0; v < 8; ++v) S[h][rl + v][cl] = acc[v];
  }
  __syncthreads();
  const int jj = tid & 15, ii = tid >> 4;
  float s[Hq];
#pragma unroll
  for (int h = 0; h < Hq; ++h) s[h] = S[h][ii][jj];
  const int ig = i0 + ii, jg = j0 + jj;
  const bool masked = (jg > ig + Mq);          // triu(diag = j-n+1): mem kv always visible
  float* outp = Dmix + ((size_t)(b * Hq) * Nq + ig) * Jq + jg;
#pragma unroll
  for (int kk = 0; kk < Hq; ++kk) {
    float d = 0.f;
#pragma unroll
    for (int h = 0; h < Hq; ++h) d = fmaf(s[h], Pp[h][kk], d);
    d *= SCALEq;
    if (masked) d = MASKVAL;
    outp[(size_t)kk * Nq * Jq] = d;
  }
}

// ---------------- Kernel 3: exact top-64 threshold + softmax (one wave per row) ----------------
__global__ void k_topk(const float* __restrict__ Dmix, _Float16* __restrict__ P_h) {
  const int lane = threadIdx.x & 31;
  const int row = blockIdx.x * 8 + (threadIdx.x >> 5);   // (b*H+kk)*N + i
  const float* __restrict__ Dr = Dmix + (size_t)row * Jq;
  unsigned keys[65];                 // register-resident row: 65 values/lane, stride 32
#pragma unroll
  for (int e = 0; e < 64; ++e) keys[e] = f2k(Dr[lane + 32 * e]);
  keys[64] = (lane < 16) ? f2k(Dr[lane + 2048]) : f2k(MASKVAL);  // tail: J=2064
  // row max
  unsigned mk = 0u;
#pragma unroll
  for (int e = 0; e < 65; ++e) mk = (keys[e] > mk) ? keys[e] : mk;
  mk = wred_max_u(mk);
  // radix bisection for exact 64-th largest key (max t with count(>=t) >= 64)
  unsigned ans = 0u;
  for (int bit = 31; bit >= 0; --bit) {
    unsigned cand = ans | (1u << bit);
    int c = 0;
#pragma unroll
    for (int e = 0; e < 65; ++e) c += (keys[e] >= cand) ? 1 : 0;
    c = wred_add_i(c);
    if (c >= TOPKq) ans = cand;
  }
  const float rmax = k2f(mk);
  float ssum = 0.f;
#pragma unroll
  for (int e = 0; e < 65; ++e) {
    bool valid = (e < 64) || (lane < 16);
    float p = (valid && keys[e] >= ans) ? __expf(k2f(keys[e]) - rmax) : 0.f;
    ssum += p;
  }
  ssum = wred_add_f(ssum);
  const float inv = 1.f / ssum;
  _Float16* __restrict__ Pr = P_h + (size_t)row * Jq;
#pragma unroll
  for (int e = 0; e < 65; ++e) {
    int j = lane + 32 * e;
    if (j < Jq) {
      float p = (keys[e] >= ans) ? __expf(k2f(keys[e]) - rmax) : 0.f;
      Pr[j] = (_Float16)(p * inv);
    }
  }
}

// ---------------- Kernel 4: PV GEMM (WMMA), O[b,i,h*64+d] ----------------
__global__ void k_pv(const _Float16* __restrict__ P_h, const _Float16* __restrict__ v_t,
                     _Float16* __restrict__ O_h) {
  const int b = blockIdx.z, h = blockIdx.y, i0 = blockIdx.x << 4;
  const int lane = threadIdx.x & 31;
  const _Float16* A  = P_h + ((size_t)(b * Hq + h) * Nq + i0) * Jq;
  const _Float16* Bb = v_t + ((size_t)(b * Hq + h) * DHq) * Jq;  // [d][j], ld=J
  v8f acc[4] = {};
  for (int k0 = 0; k0 < 2048; k0 += 32) {
    v16h a = load_a16(A, Jq, k0);
#pragma unroll
    for (int t = 0; t < 4; ++t)
      acc[t] = wmma32(a, load_b16(Bb + (size_t)(t * 16) * Jq, Jq, k0), acc[t]);
  }
  { // tail j = 2048..2063 (16 valid of 32)
    v16h a = load_a16_guard(A, Jq, 2048, Jq);
#pragma unroll
    for (int t = 0; t < 4; ++t)
      acc[t] = wmma32(a, load_b16_guard(Bb + (size_t)(t * 16) * Jq, Jq, 2048, Jq), acc[t]);
  }
  const int rl = (lane >> 4) << 3, cl = lane & 15;
#pragma unroll
  for (int t = 0; t < 4; ++t) {
#pragma unroll
    for (int v = 0; v < 8; ++v) {
      int n = i0 + rl + v;
      int d = t * 16 + cl;
      O_h[((size_t)(b * Nq) + n) * (Hq * DHq) + h * DHq + d] = (_Float16)acc[t][v];
    }
  }
}

// ---------------- Kernel 5: output GEMM (WMMA) + bias ----------------
__global__ void k_out(const _Float16* __restrict__ O_h, const _Float16* __restrict__ WoutT,
                      const float* __restrict__ bout, float* __restrict__ out) {
  const int lane = threadIdx.x & 31;
  const int row0 = blockIdx.y << 5, col0 = blockIdx.x << 6;
  const _Float16* A0 = O_h + (size_t)row0 * (Hq * DHq);
  const _Float16* A1 = A0 + (size_t)16 * (Hq * DHq);
  v8f acc[2][4] = {};
  for (int k0 = 0; k0 < Hq * DHq; k0 += 32) {
    v16h a0 = load_a16(A0, Hq * DHq, k0);
    v16h a1 = load_a16(A1, Hq * DHq, k0);
#pragma unroll
    for (int t = 0; t < 4; ++t) {
      v16h b = load_b16(WoutT + (size_t)(col0 + 16 * t) * (Hq * DHq), Hq * DHq, k0);
      acc[0][t] = wmma32(a0, b, acc[0][t]);
      acc[1][t] = wmma32(a1, b, acc[1][t]);
    }
  }
  const int rl = (lane >> 4) << 3, cl = lane & 15;
#pragma unroll
  for (int rf = 0; rf < 2; ++rf) {
#pragma unroll
    for (int t = 0; t < 4; ++t) {
#pragma unroll
      for (int v = 0; v < 8; ++v) {
        int row = row0 + 16 * rf + rl + v;
        int col = col0 + 16 * t + cl;
        out[(size_t)row * DIMq + col] = acc[rf][t][v] + bout[col];
      }
    }
  }
}

// ---------------- launch ----------------
extern "C" void kernel_launch(void* const* d_in, const int* in_sizes, int n_in,
                              void* d_out, int out_size, void* d_ws, size_t ws_size,
                              hipStream_t stream) {
  (void)in_sizes; (void)n_in; (void)out_size; (void)ws_size;
  const float* x        = (const float*)d_in[0];
  const float* Wq       = (const float*)d_in[1];
  const float* Wkv      = (const float*)d_in[2];
  const float* pre_proj = (const float*)d_in[3];
  const float* mem_k    = (const float*)d_in[4];
  const float* mem_v    = (const float*)d_in[5];
  const float* Wout     = (const float*)d_in[6];
  const float* bout     = (const float*)d_in[7];
  float* out = (float*)d_out;

  char* ws = (char*)d_ws;
  size_t off = 0;
  auto take = [&](size_t bytes) -> char* {
    char* p = ws + off;
    off += (bytes + 255) & ~(size_t)255;
    return p;
  };
  _Float16* x_h   = (_Float16*)take((size_t)Bq * Nq * DIMq * 2);
  _Float16* WqkvT = (_Float16*)take((size_t)3 * Hq * DHq * DIMq * 2);
  _Float16* WoutT = (_Float16*)take((size_t)Hq * DHq * DIMq * 2);
  _Float16* q_h   = (_Float16*)take((size_t)Bq * Hq * Nq * DHq * 2);
  _Float16* k_h   = (_Float16*)take((size_t)Bq * Hq * Jq * DHq * 2);
  _Float16* v_t   = (_Float16*)take((size_t)Bq * Hq * DHq * Jq * 2);
  float*    Dmix  = (float*)   take((size_t)Bq * Hq * Nq * Jq * 4);
  _Float16* P_h   = (_Float16*)take((size_t)Bq * Hq * Nq * Jq * 2);
  _Float16* O_h   = (_Float16*)take((size_t)Bq * Nq * Hq * DHq * 2);

  k_cvt<<<dim3((Bq * Nq * DIMq) / 256), 256, 0, stream>>>(x, x_h, Bq * Nq * DIMq);
  k_pack_wqkv<<<dim3((3072 * 1024) / 256), 256, 0, stream>>>(Wq, Wkv, WqkvT);
  k_pack_wout<<<dim3((1024 * 1024) / 256), 256, 0, stream>>>(Wout, WoutT);
  k_init_mem<<<dim3((Bq * Hq * Mq * DHq) / 256), 256, 0, stream>>>(mem_k, mem_v, k_h, v_t);

  k_gemm_qkv<<<dim3(3072 / 64, (Bq * Nq) / 32), 32, 0, stream>>>(x_h, WqkvT, q_h, k_h, v_t);
  k_dots_mix<<<dim3(Jq / 16, Nq / 16, Bq), 256, 0, stream>>>(q_h, k_h, pre_proj, Dmix);
  k_topk<<<dim3((Bq * Hq * Nq) / 8), 256, 0, stream>>>(Dmix, P_h);
  k_pv<<<dim3(Nq / 16, Hq, Bq), 32, 0, stream>>>(P_h, v_t, O_h);
  k_out<<<dim3(DIMq / 64, (Bq * Nq) / 32), 32, 0, stream>>>(O_h, WoutT, bout, out);
}
